// RoIBridge_52355651338246
// MI455X (gfx1250) — matrix operand order repaired and checked by reading, not compiled, not run on hardware
//
#include <hip/hip_runtime.h>
#include <hip/hip_bf16.h>
#include <stdint.h>

// ---------------------------------------------------------------------------
// MI455X (gfx1250) implementation.
//  N=32768 rows, feat = [poe 256 | vis 2048 | glove 300] (2604 -> pad 2624),
//  out = relu(relu(feat@W1+b1)@W2+b2)  [N,512] fp32.
//  bf16 WMMA GEMMs (v_wmma_f32_16x16x32_bf16), double-buffered LDS tiles,
//  staged via GLOBAL_LOAD_ASYNC_TO_LDS_B128 (direct memory->LDS,
//  ASYNCcnt-tracked, no VGPR staging).
// ---------------------------------------------------------------------------

typedef __attribute__((ext_vector_type(16))) __bf16 v16bf;
typedef __attribute__((ext_vector_type(8)))  __bf16 v8bf;
typedef __attribute__((ext_vector_type(8)))  float  v8f;
typedef int v4i_vec __attribute__((vector_size(16)));   // gcc-style, matches builtin

#define N_ROWS 32768
#define OBJN   256
#define DVIS   2048
#define K1     2604
#define K1PAD  2624      /* 82 * 32 */
#define NC1    1024
#define NC2    512
#define KROI   16384

// ---- async global->LDS support (gfx1250), guarded so either path compiles ---
#if defined(__has_builtin)
#  if __has_builtin(__builtin_amdgcn_global_load_async_to_lds_b128)
#    define HAVE_ASYNC_LDS 1
#  endif
#endif
#ifndef HAVE_ASYNC_LDS
#  define HAVE_ASYNC_LDS 0
#endif

#if HAVE_ASYNC_LDS
__device__ __forceinline__ void async_cp16(const __bf16* g, __bf16* l) {
    __builtin_amdgcn_global_load_async_to_lds_b128(
        (__attribute__((address_space(1))) v4i_vec*)g,
        (__attribute__((address_space(3))) v4i_vec*)l,
        0, 0);
}
__device__ __forceinline__ void wait_async0() {
#if __has_builtin(__builtin_amdgcn_s_wait_asynccnt)
    __builtin_amdgcn_s_wait_asynccnt(0);
#else
    asm volatile("s_wait_asynccnt 0x0" ::: "memory");
#endif
}
#endif

__device__ __forceinline__ __bf16 f2bf(float f) {
    union { float f; uint32_t u; } v; v.f = f;
    uint32_t r = v.u + 0x7FFFu + ((v.u >> 16) & 1u);   // round-to-nearest-even
    union { uint16_t u; __bf16 b; } o; o.u = (uint16_t)(r >> 16);
    return o.b;
}

// ---------------------------------------------------------------------------
// Prep kernels
// ---------------------------------------------------------------------------
__global__ void init_inv_kernel(int* __restrict__ inv) {
    inv[blockIdx.x * 256 + threadIdx.x] = -1;
}

__global__ void scatter_inv_kernel(const int* __restrict__ addin,
                                   int* __restrict__ inv) {
    int k = blockIdx.x * 256 + threadIdx.x;   // grid sized exactly to KROI
    inv[addin[k]] = k;
}

// W1 [2604,1024] fp32 -> W1T [1024,2624] bf16 (K-major rows, zero pad)
__global__ void w1t_kernel(const float* __restrict__ W1,
                           unsigned short* __restrict__ W1T) {
    int e = blockIdx.x * 256 + threadIdx.x;       // 1024*2624 elems
    int c = e / K1PAD;
    int k = e - c * K1PAD;
    float v = (k < K1) ? W1[(size_t)k * NC1 + c] : 0.0f;
    ((__bf16*)W1T)[(size_t)c * K1PAD + k] = f2bf(v);
}

// W2 [1024,512] fp32 -> W2T [512,1024] bf16
__global__ void w2t_kernel(const float* __restrict__ W2,
                           unsigned short* __restrict__ W2T) {
    int e = blockIdx.x * 256 + threadIdx.x;       // 512*1024 elems
    int c = e >> 10;
    int k = e & 1023;
    ((__bf16*)W2T)[(size_t)c * 1024 + k] = f2bf(W2[(size_t)k * NC2 + c]);
}

// ---------------------------------------------------------------------------
// Fused feature assembly: A_bf16[n, 0:2624] = [poe*mask | vis | glove | 0pad]
// One block per row.
// ---------------------------------------------------------------------------
__global__ __launch_bounds__(256)
void assemble_kernel(const float* __restrict__ roi,
                     const float* __restrict__ bboxs,
                     const float* __restrict__ pos_table,
                     const float* __restrict__ glove,
                     const int*   __restrict__ inv,
                     unsigned short* __restrict__ Aout) {
    const int n = blockIdx.x;
    const int r = inv[n];                 // RoI slot or -1
    const float mask = (r >= 0) ? 1.0f : 0.0f;

    int idx[4];
#pragma unroll
    for (int j = 0; j < 4; ++j) {
        float v = bboxs[n * 4 + j] * 224.0f;
        v = fminf(fmaxf(v, 0.0f), 224.0f);
        idx[j] = (int)v;                  // truncation matches astype(int32)
    }

    __bf16* A = (__bf16*)Aout + (size_t)n * K1PAD;
    const int obj = n & (OBJN - 1);

    for (int col = threadIdx.x; col < K1PAD; col += 256) {
        float v;
        if (col < 256) {                              // PoE
            int j = col >> 6, d = col & 63;
            v = pos_table[idx[j] * 64 + d] * mask;
        } else if (col < 256 + DVIS) {                // RoI visual
            v = (r >= 0) ? roi[(size_t)r * DVIS + (col - 256)] : 0.0f;
        } else if (col < K1) {                        // GloVe tile
            v = glove[obj * 300 + (col - 2304)];
        } else {                                      // K pad
            v = 0.0f;
        }
        A[col] = f2bf(v);
    }
}

// ---------------------------------------------------------------------------
// WMMA GEMM:  C[row,col] = relu( A[row,:] . Bt[col,:] + bias[col] )
//   A  : [32768, KDIM] bf16 row-major
//   Bt : [NCOLS, KDIM] bf16 row-major (weight transposed)
//   block = 256 threads = 8 waves; block tile 128x128; wave tile 32x64
//   (2x4 grid of 16x16 f32 accumulators); K step 32; double-buffered LDS.
// ---------------------------------------------------------------------------
template<int KDIM, int NCOLS, bool OUT_BF16>
__global__ __launch_bounds__(256)
void gemm_kernel(const unsigned short* __restrict__ Ain,
                 const unsigned short* __restrict__ Btin,
                 const float* __restrict__ bias,
                 void* __restrict__ Cout) {
    constexpr int LDT = 40;                    // padded LDS row stride (bf16)
    __shared__ __align__(16) __bf16 sA[2][128 * LDT];
    __shared__ __align__(16) __bf16 sB[2][128 * LDT];

    const __bf16* A  = (const __bf16*)Ain;
    const __bf16* Bt = (const __bf16*)Btin;

    const int tid  = threadIdx.x;
    const int lane = tid & 31;
    const int wave = tid >> 5;
    const int wr   = wave & 3;                 // wave row group: 32 rows each
    const int wc   = wave >> 2;                // wave col group: 64 cols each
    const int l16  = lane & 15;
    const int lhi  = lane >> 4;                // 0 | 1

    const int rowBase = blockIdx.x * 128;
    const int colBase = blockIdx.y * 128;

    // staging: each thread moves 32B of the A tile and 32B of the B tile
    const int srow  = tid >> 1;                // 0..127
    const int skoff = (tid & 1) * 16;          // bf16 units (0 or 32B)

    const __bf16* gA = A  + (size_t)(rowBase + srow) * KDIM + skoff;
    const __bf16* gB = Bt + (size_t)(colBase + srow) * KDIM + skoff;
    __bf16* lA0 = &sA[0][srow * LDT + skoff];
    __bf16* lB0 = &sB[0][srow * LDT + skoff];
    __bf16* lA1 = &sA[1][srow * LDT + skoff];
    __bf16* lB1 = &sB[1][srow * LDT + skoff];

    constexpr int NK = KDIM / 32;

    v8f acc[2][4];
#pragma unroll
    for (int i = 0; i < 2; ++i)
#pragma unroll
        for (int j = 0; j < 4; ++j) {
            v8f z = {0.f, 0.f, 0.f, 0.f, 0.f, 0.f, 0.f, 0.f};
            acc[i][j] = z;
        }

#if HAVE_ASYNC_LDS
    // ---- stage 0 via async memory->LDS (no VGPR staging) ----
    async_cp16(gA,     lA0);
    async_cp16(gA + 8, lA0 + 8);
    async_cp16(gB,     lB0);
    async_cp16(gB + 8, lB0 + 8);
    wait_async0();
#else
    {
        uint4 ra0 = *(const uint4*)(gA);
        uint4 ra1 = *(const uint4*)(gA + 8);
        uint4 rb0 = *(const uint4*)(gB);
        uint4 rb1 = *(const uint4*)(gB + 8);
        *(uint4*)lA0 = ra0; *(uint4*)(lA0 + 8) = ra1;
        *(uint4*)lB0 = rb0; *(uint4*)(lB0 + 8) = rb1;
    }
#endif
    __syncthreads();

    for (int ks = 0; ks < NK; ++ks) {
        const int cur = ks & 1;
        __bf16* nA = cur ? lA0 : lA1;
        __bf16* nB = cur ? lB0 : lB1;

#if HAVE_ASYNC_LDS
        if (ks + 1 < NK) {                      // async prefetch into nxt buf
            const __bf16* pA = gA + (size_t)(ks + 1) * 32;
            const __bf16* pB = gB + (size_t)(ks + 1) * 32;
            async_cp16(pA,     nA);
            async_cp16(pA + 8, nA + 8);
            async_cp16(pB,     nB);
            async_cp16(pB + 8, nB + 8);
        }
#else
        uint4 ra0, ra1, rb0, rb1;
        if (ks + 1 < NK) {                      // register prefetch
            const __bf16* pA = gA + (size_t)(ks + 1) * 32;
            const __bf16* pB = gB + (size_t)(ks + 1) * 32;
            ra0 = *(const uint4*)(pA);
            ra1 = *(const uint4*)(pA + 8);
            rb0 = *(const uint4*)(pB);
            rb1 = *(const uint4*)(pB + 8);
        }
#endif

        // A fragment (ISA 16-bit A 16x32 layout):
        //   lane<16 holds K{0..7,16..23}; lane>=16 holds K{8..15,24..31}
        union Frag { v16bf v; v8bf h[2]; };
        Frag fa[2], fb[4];
#pragma unroll
        for (int tr = 0; tr < 2; ++tr) {
            const __bf16* p = &sA[cur][(wr * 32 + tr * 16 + l16) * LDT + lhi * 8];
            fa[tr].h[0] = *(const v8bf*)(p);
            fa[tr].h[1] = *(const v8bf*)(p + 16);
        }
        // B fragment: lane<16 -> col=l16, K{0..7,16..23}; lane>=16 -> K{8..15,24..31}
#pragma unroll
        for (int tc = 0; tc < 4; ++tc) {
            const __bf16* p = &sB[cur][(wc * 64 + tc * 16 + l16) * LDT + lhi * 8];
            fb[tc].h[0] = *(const v8bf*)(p);
            fb[tc].h[1] = *(const v8bf*)(p + 16);
        }

#pragma unroll
        for (int tr = 0; tr < 2; ++tr)
#pragma unroll
            for (int tc = 0; tc < 4; ++tc)
                acc[tr][tc] = __builtin_amdgcn_wmma_f32_16x16x32_bf16(
                    false, fa[tr].v, false, fb[tc].v,
                    (short)0, acc[tr][tc], false, false);

#if HAVE_ASYNC_LDS
        if (ks + 1 < NK) wait_async0();
#else
        if (ks + 1 < NK) {
            *(uint4*)nA = ra0; *(uint4*)(nA + 8) = ra1;
            *(uint4*)nB = rb0; *(uint4*)(nB + 8) = rb1;
        }
#endif
        __syncthreads();
    }

    // epilogue: bias + relu, C/D layout: VGPR i -> M = lhi*8 + i, N = l16
#pragma unroll
    for (int tc = 0; tc < 4; ++tc) {
        const int col = colBase + wc * 64 + tc * 16 + l16;
        const float bv = bias[col];
#pragma unroll
        for (int tr = 0; tr < 2; ++tr) {
            const int row0 = rowBase + wr * 32 + tr * 16 + lhi * 8;
#pragma unroll
            for (int i = 0; i < 8; ++i) {
                float v = acc[tr][tc][i] + bv;
                v = v > 0.0f ? v : 0.0f;
                if constexpr (OUT_BF16)
                    ((__bf16*)Cout)[(size_t)(row0 + i) * NCOLS + col] = f2bf(v);
                else
                    ((float*)Cout)[(size_t)(row0 + i) * NCOLS + col] = v;
            }
        }
    }
}

// ---------------------------------------------------------------------------
// Host launcher
// ---------------------------------------------------------------------------
extern "C" void kernel_launch(void* const* d_in, const int* in_sizes, int n_in,
                              void* d_out, int out_size, void* d_ws, size_t ws_size,
                              hipStream_t stream) {
    const float* roi   = (const float*)d_in[0];
    const float* bboxs = (const float*)d_in[1];
    const float* pos   = (const float*)d_in[2];
    const float* glove = (const float*)d_in[3];
    const float* W1    = (const float*)d_in[4];
    const float* b1    = (const float*)d_in[5];
    const float* W2    = (const float*)d_in[6];
    const float* b2    = (const float*)d_in[7];
    const int*   addin = (const int*)d_in[8];

    // workspace layout (256B aligned regions)
    char* ws = (char*)d_ws;
    const size_t offA   = 131072;                              // after inv
    const size_t szA    = (size_t)N_ROWS * K1PAD * 2;          // 171,966,464
    const size_t offH   = offA + szA;
    const size_t szH    = (size_t)N_ROWS * NC1 * 2;            // 67,108,864
    const size_t offW1T = offH + szH;
    const size_t szW1T  = (size_t)NC1 * K1PAD * 2;             // 5,373,952
    const size_t offW2T = offW1T + szW1T;

    int*            inv  = (int*)ws;
    unsigned short* Abf  = (unsigned short*)(ws + offA);
    unsigned short* Hbf  = (unsigned short*)(ws + offH);
    unsigned short* W1T  = (unsigned short*)(ws + offW1T);
    unsigned short* W2T  = (unsigned short*)(ws + offW2T);

    init_inv_kernel<<<N_ROWS / 256, 256, 0, stream>>>(inv);
    scatter_inv_kernel<<<KROI / 256, 256, 0, stream>>>(addin, inv);
    w1t_kernel<<<(NC1 * K1PAD) / 256, 256, 0, stream>>>(W1, W1T);
    w2t_kernel<<<(NC2 * NC1) / 256, 256, 0, stream>>>(W2, W2T);
    assemble_kernel<<<N_ROWS, 256, 0, stream>>>(roi, bboxs, pos, glove, inv, Abf);

    dim3 g1(N_ROWS / 128, NC1 / 128);   // 256 x 8
    gemm_kernel<K1PAD, NC1, true><<<g1, 256, 0, stream>>>(Abf, W1T, b1, Hbf);

    dim3 g2(N_ROWS / 128, NC2 / 128);   // 256 x 4
    gemm_kernel<NC1, NC2, false><<<g2, 256, 0, stream>>>(Hbf, W2T, b2, (float*)d_out);
}
